// GOGraphEncoder_43868795961640
// MI455X (gfx1250) — matrix operand order repaired and checked by reading, not compiled, not run on hardware
//
#include <hip/hip_runtime.h>
#include <math.h>

// ---------------- problem constants ----------------
namespace cfg {
constexpr int N   = 30000;
constexpr int E   = 90000;
constexpr int D   = 2560;
constexpr int HID = 512;
constexpr int NH  = 8;     // GAT heads
constexpr int HC  = 64;    // GAT head channels
constexpr int Q   = 200;
constexpr int AH  = 8;     // cross-attn heads
constexpr int ADH = 320;   // cross-attn head dim
constexpr int FF  = 4 * D; // 10240
}

// ---------------- WMMA vector types ----------------
typedef __bf16 bf16_t;
typedef bf16_t v16bf __attribute__((ext_vector_type(16)));
typedef bf16_t v4bf  __attribute__((ext_vector_type(4)));
typedef float  v8f   __attribute__((ext_vector_type(8)));

__device__ __forceinline__ float gelu_f(float x) {
  return 0.5f * x * (1.0f + erff(x * 0.70710678118654752440f));
}
__device__ __forceinline__ float leaky02(float x) { return x > 0.f ? x : 0.2f * x; }

// float atomic max via integer ordering trick (init must be -inf)
__device__ __forceinline__ void atomicMaxF(float* addr, float val) {
  if (val >= 0.f) atomicMax((int*)addr, __float_as_int(val));
  else            atomicMin((unsigned int*)addr, __float_as_uint(val));
}

// ---------------------------------------------------------------------------
// bf16-WMMA GEMM:  C[M,Nc] = act( alpha * A[M,K] @ B[K,Nc] + bias + res )
// REQUIRES: K % 4 == 0 (all call sites satisfy this), strides multiple of 4.
// Block tile 128x64, 256 threads = 8 waves; wave (wr, wcol) owns a 32x32 C
// patch = 4 WMMA accumulators. K stepped by 32.
//
// LDS layout is fragment-ordered so each lane fragment is 32 contiguous bytes:
//  - As[row][k] (row-major, unpermuted):
//    lane half 0 reads As[row][0..15]  = K {0..15}
//    lane half 1 reads As[row][16..31] = K {16..31}
//  - Bt[n][k] (n-major): lane half h reads Bt[n][16h..16h+15] = K {16h..16h+15}
// Since gc/gk are multiples of 4 and K % 4 == 0, a single per-vector bound
// check is exact -> no per-element fallback path.
// ---------------------------------------------------------------------------
#define BM 128
#define BN 64
#define BK 32

__global__ __launch_bounds__(256) void gemm_bf16_wmma(
    const float* __restrict__ A, long lda,
    const float* __restrict__ B, long ldb, int transB,
    const float* __restrict__ bias,            // per output column, may be null
    const float* __restrict__ res, long ldr,   // residual [M,Nc], may be null
    float* __restrict__ C, long ldc,
    int M, int K, int Nc, float alpha, int act)
{
  __shared__ __align__(16) bf16_t As[BM][BK];
  __shared__ __align__(16) bf16_t Bt[BN][BK];

  const int bm = blockIdx.y * BM;
  const int bn = blockIdx.x * BN;
  const int tid  = threadIdx.x;
  const int wave = tid >> 5;
  const int lane = tid & 31;
  const int wr   = wave >> 1;      // 0..3 : rows wr*32
  const int wcol = wave & 1;       // 0..1 : cols wcol*32
  const int hlf  = lane >> 4;
  const int l16  = lane & 15;

  v8f acc00 = {}, acc01 = {}, acc10 = {}, acc11 = {};

  for (int k0 = 0; k0 < K; k0 += BK) {
    // ---- stage A: 128x32 floats -> bf16, 4 x float4 per thread ----
    #pragma unroll
    for (int i = 0; i < 4; ++i) {
      int idx = tid + i * 256;           // 0..1023 vec4-id
      int r   = idx >> 3;                // 0..127
      int cv  = (idx & 7) * 4;           // 0,4,...,28
      int gr = bm + r, gc = k0 + cv;
      float4 v4 = make_float4(0.f, 0.f, 0.f, 0.f);
      if (gr < M && gc < K) {            // exact: gc % 4 == 0, K % 4 == 0
        v4 = *(const float4*)(A + (long)gr * lda + gc);
        if (gc + BK < K) __builtin_prefetch(A + (long)gr * lda + gc + BK, 0, 3);
      }
      int g  = cv >> 3;                  // group of 8 along K
      int pc = cv + ((g == 1) ? 0 : (g == 2) ? 0 : 0);    // no permute (contiguous K)
      v4bf t;
      t[0] = (bf16_t)v4.x; t[1] = (bf16_t)v4.y;
      t[2] = (bf16_t)v4.z; t[3] = (bf16_t)v4.w;
      *(v4bf*)&As[r][pc] = t;            // ds_store_b64
    }
    // ---- stage B: 64 n x 32 k -> Bt[n][k], 2 x vec4 tasks per thread ----
    #pragma unroll
    for (int i = 0; i < 2; ++i) {
      int idx = tid + i * 256;           // 0..511
      int n   = idx & 63;                // consecutive threads -> consecutive n
      int kv  = (idx >> 6) * 4;          // 0,4,...,28
      int gn = bn + n, gk = k0 + kv;
      v4bf t = {};
      if (!transB) {
        if (gn < Nc && gk < K) {         // exact vector guard
          const float* bp = B + (long)gk * ldb + gn;
          t[0] = (bf16_t)bp[0];
          t[1] = (bf16_t)bp[ldb];
          t[2] = (bf16_t)bp[2 * ldb];
          t[3] = (bf16_t)bp[3 * ldb];
          if (gk + BK < K) __builtin_prefetch(bp + (long)BK * ldb, 0, 3);
        }
      } else {                           // B[n*ldb + k]
        if (gn < Nc && gk < K) {
          float4 v4 = *(const float4*)(B + (long)gn * ldb + gk);
          t[0] = (bf16_t)v4.x; t[1] = (bf16_t)v4.y;
          t[2] = (bf16_t)v4.z; t[3] = (bf16_t)v4.w;
          if (gk + BK < K) __builtin_prefetch(B + (long)gn * ldb + gk + BK, 0, 3);
        }
      }
      *(v4bf*)&Bt[n][kv] = t;            // ds_store_b64
    }
    __syncthreads();

    // ---- fragments: 32 contiguous bytes each -> 2x ds_load_b128 ----
    v16bf a0 = *(const v16bf*)&As[wr * 32 +      l16][hlf * 16];
    v16bf a1 = *(const v16bf*)&As[wr * 32 + 16 + l16][hlf * 16];
    v16bf b0 = *(const v16bf*)&Bt[wcol * 32 +      l16][hlf * 16];
    v16bf b1 = *(const v16bf*)&Bt[wcol * 32 + 16 + l16][hlf * 16];

    acc00 = __builtin_amdgcn_wmma_f32_16x16x32_bf16(false, a0, false, b0, (short)0, acc00, false, false);
    acc01 = __builtin_amdgcn_wmma_f32_16x16x32_bf16(false, a0, false, b1, (short)0, acc01, false, false);
    acc10 = __builtin_amdgcn_wmma_f32_16x16x32_bf16(false, a1, false, b0, (short)0, acc10, false, false);
    acc11 = __builtin_amdgcn_wmma_f32_16x16x32_bf16(false, a1, false, b1, (short)0, acc11, false, false);
    __syncthreads();
  }

  // ---- epilogue: C/D layout VGPR r -> row sub*16 + hlf*8 + r, col l16 ----
  #pragma unroll
  for (int sub = 0; sub < 2; ++sub) {
    #pragma unroll
    for (int t = 0; t < 2; ++t) {
      v8f a = (sub == 0) ? (t == 0 ? acc00 : acc01) : (t == 0 ? acc10 : acc11);
      int cn = bn + wcol * 32 + t * 16 + l16;
      if (cn >= Nc) continue;
      float badd = bias ? bias[cn] : 0.f;
      #pragma unroll
      for (int r = 0; r < 8; ++r) {
        int cm = bm + wr * 32 + sub * 16 + hlf * 8 + r;
        if (cm >= M) continue;
        float xv = a[r] * alpha + badd;
        if (res) xv += res[(long)cm * ldr + cn];
        if (act == 1) xv = gelu_f(xv);
        C[(long)cm * ldc + cn] = xv;
      }
    }
  }
}

// ---------------- elementwise / fill ----------------
__global__ __launch_bounds__(256) void fill_kernel(float* p, long n, float v) {
  long i = (long)blockIdx.x * blockDim.x + threadIdx.x;
  if (i < n) p[i] = v;
}

__global__ __launch_bounds__(256) void add_emb_kernel(
    const float* __restrict__ x, const float* __restrict__ emb,
    float* __restrict__ out, long n, int d) {
  long i = (long)blockIdx.x * blockDim.x + threadIdx.x;
  if (i < n) out[i] = x[i] + emb[i % d];
}

// ---------------- GAT pieces ----------------
__global__ __launch_bounds__(256) void gat_esed_kernel(
    const float* __restrict__ hh, const float* __restrict__ a_s,
    const float* __restrict__ a_d, float* __restrict__ es,
    float* __restrict__ ed, int n) {
  int idx = blockIdx.x * blockDim.x + threadIdx.x;
  if (idx >= n * cfg::NH) return;
  int node = idx >> 3, hd = idx & 7;
  const float* hr = hh + (long)node * cfg::HID + hd * cfg::HC;
  const float* as = a_s + hd * cfg::HC;
  const float* ad = a_d + hd * cfg::HC;
  float s = 0.f, d = 0.f;
  #pragma unroll 8
  for (int c = 0; c < cfg::HC; ++c) { float h = hr[c]; s += h * as[c]; d += h * ad[c]; }
  es[idx] = s; ed[idx] = d;
}

__device__ __forceinline__ void edge_decode(const int* __restrict__ ei, int e,
                                            int& src, int& dst) {
  if (e < cfg::E) { src = ei[e]; dst = ei[cfg::E + e]; }
  else            { src = dst = e - cfg::E; }   // self loops
}

__global__ __launch_bounds__(256) void gat_edge_max_kernel(
    const int* __restrict__ ei, const float* __restrict__ es,
    const float* __restrict__ ed, float* __restrict__ mbuf) {
  int t = blockIdx.x * blockDim.x + threadIdx.x;
  if (t >= (cfg::E + cfg::N) * cfg::NH) return;
  int e = t >> 3, hd = t & 7, src, dst;
  edge_decode(ei, e, src, dst);
  float lg = leaky02(es[src * cfg::NH + hd] + ed[dst * cfg::NH + hd]);
  atomicMaxF(&mbuf[dst * cfg::NH + hd], lg);
}

__global__ __launch_bounds__(256) void gat_edge_sum_kernel(
    const int* __restrict__ ei, const float* __restrict__ es,
    const float* __restrict__ ed, const float* __restrict__ mbuf,
    float* __restrict__ denb) {
  int t = blockIdx.x * blockDim.x + threadIdx.x;
  if (t >= (cfg::E + cfg::N) * cfg::NH) return;
  int e = t >> 3, hd = t & 7, src, dst;
  edge_decode(ei, e, src, dst);
  float lg = leaky02(es[src * cfg::NH + hd] + ed[dst * cfg::NH + hd]);
  float ex = __expf(lg - mbuf[dst * cfg::NH + hd]);
  atomicAdd(&denb[dst * cfg::NH + hd], ex);
}

__global__ __launch_bounds__(256) void gat_edge_agg_kernel(
    const int* __restrict__ ei, const float* __restrict__ es,
    const float* __restrict__ ed, const float* __restrict__ mbuf,
    const float* __restrict__ denb, const float* __restrict__ hh,
    float* __restrict__ agg) {
  int t = blockIdx.x * blockDim.x + threadIdx.x;
  if (t >= (cfg::E + cfg::N) * cfg::NH) return;
  int e = t >> 3, hd = t & 7, src, dst;
  edge_decode(ei, e, src, dst);
  float lg = leaky02(es[src * cfg::NH + hd] + ed[dst * cfg::NH + hd]);
  float ex = __expf(lg - mbuf[dst * cfg::NH + hd]);
  float alpha = ex / (denb[dst * cfg::NH + hd] + 1e-16f);
  const float* hr = hh + (long)src * cfg::HID + hd * cfg::HC;
  float* orow = agg + (long)dst * cfg::HID + hd * cfg::HC;
  #pragma unroll 4
  for (int c = 0; c < cfg::HC; ++c) atomicAdd(&orow[c], hr[c] * alpha);
}

__global__ __launch_bounds__(256) void gat_finalize_kernel(
    float* __restrict__ agg, const float* __restrict__ bias, long n, int do_elu) {
  long i = (long)blockIdx.x * blockDim.x + threadIdx.x;
  if (i >= n) return;
  int col = (int)(i % cfg::HID);
  float x = agg[i] + bias[col];
  if (do_elu) x = x > 0.f ? x : (__expf(x) - 1.f);
  agg[i] = x;
}

// ---------------- row softmax / layernorm ----------------
__global__ __launch_bounds__(256) void softmax_rows_kernel(float* __restrict__ s, int cols) {
  __shared__ float red[256];
  const int tid = threadIdx.x;
  float* p = s + (long)blockIdx.x * cols;
  float m = -INFINITY;
  for (int i = tid; i < cols; i += 256) m = fmaxf(m, p[i]);
  red[tid] = m; __syncthreads();
  for (int o = 128; o > 0; o >>= 1) { if (tid < o) red[tid] = fmaxf(red[tid], red[tid + o]); __syncthreads(); }
  m = red[0]; __syncthreads();
  float sum = 0.f;
  for (int i = tid; i < cols; i += 256) { float e = __expf(p[i] - m); p[i] = e; sum += e; }
  red[tid] = sum; __syncthreads();
  for (int o = 128; o > 0; o >>= 1) { if (tid < o) red[tid] += red[tid + o]; __syncthreads(); }
  float inv = 1.f / red[0];
  for (int i = tid; i < cols; i += 256) p[i] *= inv;
}

__global__ __launch_bounds__(256) void layernorm_kernel(
    const float* __restrict__ in, const float* __restrict__ g,
    const float* __restrict__ b, float* __restrict__ out, int cols) {
  __shared__ float red[256];
  const int tid = threadIdx.x;
  const float* p = in + (long)blockIdx.x * cols;
  float s = 0.f;
  for (int i = tid; i < cols; i += 256) s += p[i];
  red[tid] = s; __syncthreads();
  for (int o = 128; o > 0; o >>= 1) { if (tid < o) red[tid] += red[tid + o]; __syncthreads(); }
  float mean = red[0] / cols; __syncthreads();
  float v = 0.f;
  for (int i = tid; i < cols; i += 256) { float d = p[i] - mean; v += d * d; }
  red[tid] = v; __syncthreads();
  for (int o = 128; o > 0; o >>= 1) { if (tid < o) red[tid] += red[tid + o]; __syncthreads(); }
  float inv = rsqrtf(red[0] / cols + 1e-5f);
  float* q = out + (long)blockIdx.x * cols;
  for (int i = tid; i < cols; i += 256) q[i] = (p[i] - mean) * inv * g[i] + b[i];
}

// ---------------------------------------------------------------------------
extern "C" void kernel_launch(void* const* d_in, const int* in_sizes, int n_in,
                              void* d_out, int out_size, void* d_ws, size_t ws_size,
                              hipStream_t stream) {
  using namespace cfg;
  const float* x        = (const float*)d_in[0];
  const int*   ei       = (const int*)  d_in[1];
  const float* ns_emb   = (const float*)d_in[2];
  const float* gat_w0   = (const float*)d_in[3];
  const float* gat_wr   = (const float*)d_in[4];
  const float* a_src    = (const float*)d_in[5];
  const float* a_dst    = (const float*)d_in[6];
  const float* gat_b    = (const float*)d_in[7];
  const float* proj_w1  = (const float*)d_in[8];
  const float* proj_b1  = (const float*)d_in[9];
  const float* proj_w2  = (const float*)d_in[10];
  const float* proj_b2  = (const float*)d_in[11];
  const float* queries  = (const float*)d_in[12];
  const float* wq = (const float*)d_in[13]; const float* bq = (const float*)d_in[14];
  const float* wk = (const float*)d_in[15]; const float* bk = (const float*)d_in[16];
  const float* wv = (const float*)d_in[17]; const float* bv = (const float*)d_in[18];
  const float* wo = (const float*)d_in[19]; const float* bo = (const float*)d_in[20];
  const float* ln1_g = (const float*)d_in[21]; const float* ln1_b = (const float*)d_in[22];
  const float* ffn_w1 = (const float*)d_in[23]; const float* ffn_b1 = (const float*)d_in[24];
  const float* ffn_w2 = (const float*)d_in[25]; const float* ffn_b2 = (const float*)d_in[26];
  const float* ln2_g = (const float*)d_in[27]; const float* ln2_b = (const float*)d_in[28];
  float* out = (float*)d_out;

  // ---- workspace carve-up (floats) ----
  float* ws = (float*)d_ws;
  const size_t ND  = (size_t)N * D;
  const size_t NHd = (size_t)N * HID;
  const size_t NNH = (size_t)N * NH;
  float* buf_h = ws;                 // [N,D] h0, later proj output h
  float* kbuf  = buf_h + ND;         // [N,D]
  float* vbuf  = kbuf + ND;          // [N,D]
  float* T1    = vbuf + ND;          // [N,HID] hh
  float* T2    = T1 + NHd;           // [N,HID] layer in/out (agg)
  float* esb   = T2 + NHd;           // [N,NH]
  float* edb   = esb + NNH;
  float* mbuf  = edb + NNH;
  float* denb  = mbuf + NNH;
  float* scr   = denb + NNH;         // [Q,N] per-head scores
  float* qb    = scr + (size_t)Q * N;    // [Q,D]
  float* ctxb  = qb + (size_t)Q * D;     // [Q,D]
  float* attb  = ctxb + (size_t)Q * D;   // [Q,D]
  float* attln = attb + (size_t)Q * D;   // [Q,D]
  float* ffh   = attln + (size_t)Q * D;  // [Q,FF]
  float* ffo   = ffh + (size_t)Q * FF;   // [Q,D]

  auto gemm = [&](const float* A, long lda, const float* B, long ldb, int tB,
                  const float* bias, const float* res, long ldr,
                  float* C, long ldc, int M, int K, int Nc, float alpha, int act) {
    dim3 g((Nc + BN - 1) / BN, (M + BM - 1) / BM);
    gemm_bf16_wmma<<<g, 256, 0, stream>>>(A, lda, B, ldb, tB, bias, res, ldr,
                                          C, ldc, M, K, Nc, alpha, act);
  };
  auto fill = [&](float* p, long n, float v) {
    fill_kernel<<<(int)((n + 255) / 256), 256, 0, stream>>>(p, n, v);
  };

  const int ET8 = (E + N) * NH;           // edge-parallel work items
  const int egrid = (ET8 + 255) / 256;

  // h0 = x + ns_emb
  add_emb_kernel<<<(int)((ND + 255) / 256), 256, 0, stream>>>(x, ns_emb, buf_h, (long)ND, D);

  // ---- 3 GAT layers ----
  for (int l = 0; l < 3; ++l) {
    const float* hin = (l == 0) ? buf_h : T2;
    long lda = (l == 0) ? D : HID;
    const float* W = (l == 0) ? gat_w0 : (gat_wr + (size_t)(l - 1) * HID * HID);
    int K = (l == 0) ? D : HID;
    gemm(hin, lda, W, HID, 0, nullptr, nullptr, 0, T1, HID, N, K, HID, 1.f, 0);
    gat_esed_kernel<<<(N * NH + 255) / 256, 256, 0, stream>>>(
        T1, a_src + (size_t)l * NH * HC, a_dst + (size_t)l * NH * HC, esb, edb, N);
    fill(mbuf, NNH, -INFINITY);
    fill(denb, NNH, 0.f);
    fill(T2, NHd, 0.f);
    gat_edge_max_kernel<<<egrid, 256, 0, stream>>>(ei, esb, edb, mbuf);
    gat_edge_sum_kernel<<<egrid, 256, 0, stream>>>(ei, esb, edb, mbuf, denb);
    gat_edge_agg_kernel<<<egrid, 256, 0, stream>>>(ei, esb, edb, mbuf, denb, T1, T2);
    gat_finalize_kernel<<<(int)((NHd + 255) / 256), 256, 0, stream>>>(
        T2, gat_b + (size_t)l * HID, (long)NHd, l < 2 ? 1 : 0);
  }

  // ---- projection: h = GELU(T2@W1+b1)@W2+b2 -> buf_h [N,D] ----
  gemm(T2, HID, proj_w1, HID, 0, proj_b1, nullptr, 0, T1, HID, N, HID, HID, 1.f, 1);
  gemm(T1, HID, proj_w2, D, 0, proj_b2, nullptr, 0, buf_h, D, N, HID, D, 1.f, 0);

  // ---- Q/K/V projections ----
  gemm(queries, D, wq, D, 0, bq, nullptr, 0, qb, D, Q, D, D, 1.f, 0);
  gemm(buf_h, D, wk, D, 0, bk, nullptr, 0, kbuf, D, N, D, D, 1.f, 0);
  gemm(buf_h, D, wv, D, 0, bv, nullptr, 0, vbuf, D, N, D, D, 1.f, 0);

  // ---- cross attention, one head at a time (scores scratch = Q x N) ----
  const float scale = 0.05590169943749474f;   // 1/sqrt(320)
  for (int h = 0; h < AH; ++h) {
    // scores[q,n] = scale * Qh[q,:] . Kh[n,:]
    gemm(qb + (size_t)h * ADH, D, kbuf + (size_t)h * ADH, D, 1,
         nullptr, nullptr, 0, scr, N, Q, ADH, N, scale, 0);
    softmax_rows_kernel<<<Q, 256, 0, stream>>>(scr, N);
    // ctx_h[q,:] = attn[q,:] @ Vh
    gemm(scr, N, vbuf + (size_t)h * ADH, D, 0,
         nullptr, nullptr, 0, ctxb + (size_t)h * ADH, D, Q, N, ADH, 1.f, 0);
  }

  // ---- attended = LN1(queries + ctx@wo + bo) ----
  gemm(ctxb, D, wo, D, 0, bo, queries, D, attb, D, Q, D, D, 1.f, 0);
  layernorm_kernel<<<Q, 256, 0, stream>>>(attb, ln1_g, ln1_b, attln, D);

  // ---- FFN + LN2 -> out ----
  gemm(attln, D, ffn_w1, FF, 0, ffn_b1, nullptr, 0, ffh, FF, Q, D, FF, 1.f, 1);
  gemm(ffh, FF, ffn_w2, D, 0, ffn_b2, attln, D, ffo, D, Q, FF, D, 1.f, 0);
  layernorm_kernel<<<Q, 256, 0, stream>>>(ffo, ln2_g, ln2_b, out, D);
  (void)in_sizes; (void)n_in; (void)out_size; (void)ws_size;
}
